// Attention_7352984011167
// MI455X (gfx1250) — matrix operand order repaired and checked by reading, not compiled, not run on hardware
//
#include <hip/hip_runtime.h>
#include <hip/hip_bf16.h>

// ---------------------------------------------------------------------------
// MViTv2/ViTDet decomposed-rel-pos attention for MI455X (gfx1250, wave32).
// B=8, H=W=32, S=1024, DIM=768, NH=12, HD=64.
// f16 WMMA for all GEMMs + flash attention. LDS layouts are arranged so every
// WMMA operand fragment is 32 contiguous bytes per lane -> 2x ds_load_b128.
//   A-operand tiles: row-major (lane = M, K contiguous, same k order as B)
//   B-operand tiles: N-major (lane = N, K contiguous)
// ---------------------------------------------------------------------------

typedef __attribute__((ext_vector_type(16))) _Float16 v16h;
typedef __attribute__((ext_vector_type(8)))  _Float16 v8h;
typedef __attribute__((ext_vector_type(8)))  float    v8f;
typedef __attribute__((ext_vector_type(4)))  float    vf4;

#define DIMC   768
#define SEQ    1024
#define NHEAD  12
#define HDIM   64
#define BATCH  8
#define BH     (BATCH * NHEAD)   // 96
#define SCALE  0.125f            // 64^-0.5

__device__ __forceinline__ v8f wmma_f16(v16h a, v16h b, v8f c) {
  return __builtin_amdgcn_wmma_f32_16x16x32_f16(
      /*neg_a=*/false, a, /*neg_b=*/false, b,
      /*c_mod=*/(short)0, c, /*reuse_a=*/false, /*reuse_b=*/false);
}

// 16 contiguous halves -> v16h via two b128 LDS loads.
__device__ __forceinline__ v16h frag16(const _Float16* p) {
  const v8h lo = *(const v8h*)p;
  const v8h hi = *(const v8h*)(p + 8);
  return __builtin_shufflevector(lo, hi, 0, 1, 2, 3, 4, 5, 6, 7, 8, 9, 10, 11,
                                 12, 13, 14, 15);
}

// Both A (row-major) and B (N-major) fragments use identical address
// math: lane (m|n) = lane&15 picks the row, lane>>4 picks the 16-half chunk.
__device__ __forceinline__ v16h load_frag(const _Float16* tile, int ld) {
  const int lane = threadIdx.x & 31;
  return frag16(tile + (lane & 15) * ld + ((lane >> 4) << 4));
}

// fp32 x8 -> f16 x8
__device__ __forceinline__ v8h cvt8(vf4 a, vf4 b) {
  v8h h;
#pragma unroll
  for (int i = 0; i < 4; ++i) {
    h[i]     = (_Float16)a[i];
    h[i + 4] = (_Float16)b[i];
  }
  return h;
}

// 8-wide K-chunk order for A tiles: identity (contiguous k, matches B tiles).
__device__ __forceinline__ int permblk(int kb) {
  return kb;
}

// ---------------------------------------------------------------------------
// Kernel 1: qkv = x @ qkv_w^T + qkv_b ; scatter into f16 Q[s][hd], K[s][hd],
// V^T[hd][s]. M=8192, N=2304, K=768. Tile 128x128x32, 8 waves.
// ---------------------------------------------------------------------------
__global__ __launch_bounds__(256) void qkv_gemm_kernel(
    const float* __restrict__ x, const float* __restrict__ qkv_w,
    const float* __restrict__ qkv_b,
    _Float16* __restrict__ qbuf, _Float16* __restrict__ kbuf,
    _Float16* __restrict__ vT) {
  __shared__ __align__(16) _Float16 As[128][40];  // [m][k], pad->80B rows
  __shared__ __align__(16) _Float16 Bs[128][40];  // [n][k] = w[n][k] verbatim

  const int tid  = threadIdx.x;
  const int wave = tid >> 5;
  const int lane = tid & 31;
  const int bm   = blockIdx.y * 128;
  const int bn   = blockIdx.x * 128;
  const int wm   = (wave & 3) * 32;
  const int wn   = (wave >> 2) * 64;

  v8f acc[2][4];
#pragma unroll
  for (int i = 0; i < 2; ++i)
#pragma unroll
    for (int j = 0; j < 4; ++j)
      acc[i][j] = (v8f){0.f, 0.f, 0.f, 0.f, 0.f, 0.f, 0.f, 0.f};

  for (int kt = 0; kt < DIMC; kt += 32) {
    // Stage A (k contiguous) and B (verbatim): 512 8-half chunks each, 2 per thread.
#pragma unroll
    for (int c = 0; c < 2; ++c) {
      const int idx = tid * 2 + c;     // 0..511
      const int row = idx >> 2;        // 0..127
      const int kb  = idx & 3;         // 8-wide k chunk
      const float* xp = x + (size_t)(bm + row) * DIMC + kt + kb * 8;
      const float* wp = qkv_w + (size_t)(bn + row) * DIMC + kt + kb * 8;
      if (kt + 32 < DIMC) {
        __builtin_prefetch(xp + 32, 0, 0);
        __builtin_prefetch(wp + 32, 0, 0);
      }
      *(v8h*)(&As[row][permblk(kb) * 8]) =
          cvt8(*(const vf4*)xp, *(const vf4*)(xp + 4));
      *(v8h*)(&Bs[row][kb * 8]) =
          cvt8(*(const vf4*)wp, *(const vf4*)(wp + 4));
    }
    __syncthreads();

    const v16h a0 = load_frag(&As[wm][0], 40);
    const v16h a1 = load_frag(&As[wm + 16][0], 40);
#pragma unroll
    for (int j = 0; j < 4; ++j) {
      const v16h b = load_frag(&Bs[wn + j * 16][0], 40);
      acc[0][j] = wmma_f16(a0, b, acc[0][j]);
      acc[1][j] = wmma_f16(a1, b, acc[1][j]);
    }
    __syncthreads();
  }

  // Epilogue: bias + scatter to q / k (row-major) and v (transposed), f16.
  const int nl = lane & 15;
  const int mb = (lane >> 4) << 3;
#pragma unroll
  for (int i = 0; i < 2; ++i)
#pragma unroll
    for (int j = 0; j < 4; ++j)
#pragma unroll
      for (int r = 0; r < 8; ++r) {
        const int gm = bm + wm + i * 16 + mb + r;
        const int gn = bn + wn + j * 16 + nl;
        const float val = acc[i][j][r] + qkv_b[gn];
        const int which = gn / DIMC;   // 0=q 1=k 2=v
        const int d     = gn % DIMC;
        const int head  = d >> 6;
        const int hd    = d & 63;
        const int b     = gm >> 10;
        const int s     = gm & 1023;
        const int bh    = b * NHEAD + head;
        if (which == 0)
          qbuf[((size_t)bh * SEQ + s) * HDIM + hd] = (_Float16)val;
        else if (which == 1)
          kbuf[((size_t)bh * SEQ + s) * HDIM + hd] = (_Float16)val;
        else
          vT[((size_t)bh * HDIM + hd) * SEQ + s] = (_Float16)val;
      }
}

// ---------------------------------------------------------------------------
// Kernel 2: rel_h[bh,s,kh] = sum_c q[bh,s,c]*rel_pos_h[h-kh+31,c] (ditto w).
// ---------------------------------------------------------------------------
__global__ __launch_bounds__(256) void rel_bias_kernel(
    const _Float16* __restrict__ qbuf,
    const float* __restrict__ rel_pos_h, const float* __restrict__ rel_pos_w,
    float* __restrict__ relh, float* __restrict__ relw) {
  const int idx = blockIdx.x * 4 + (threadIdx.x >> 6);  // (bh,s) flat
  const int t   = threadIdx.x & 63;
  const int bh  = idx >> 10;
  const int s   = idx & 1023;
  const int h   = s >> 5;
  const int w   = s & 31;
  const int k   = t & 31;
  const _Float16* qp = qbuf + ((size_t)bh * SEQ + s) * HDIM;
  const float* rp = (t < 32) ? (rel_pos_h + (size_t)(h - k + 31) * HDIM)
                             : (rel_pos_w + (size_t)(w - k + 31) * HDIM);
  float sum = 0.f;
#pragma unroll
  for (int cb = 0; cb < 8; ++cb) {
    const v8h qv  = *(const v8h*)(qp + cb * 8);
    const vf4 r0  = *(const vf4*)(rp + cb * 8);
    const vf4 r1  = *(const vf4*)(rp + cb * 8 + 4);
#pragma unroll
    for (int i = 0; i < 4; ++i)
      sum += (float)qv[i] * r0[i] + (float)qv[i + 4] * r1[i];
  }
  float* dst = (t < 32) ? relh : relw;
  dst[((size_t)bh * SEQ + s) * 32 + k] = sum;
}

// ---------------------------------------------------------------------------
// Kernel 3: flash attention. Block = 1 head x 128 query rows, 8 waves,
// 64-key tiles. S = scale*(Q K^T) + rel_h + rel_w ; online softmax ; O += P V.
// ---------------------------------------------------------------------------
__global__ __launch_bounds__(256) void flash_attn_kernel(
    const _Float16* __restrict__ qbuf, const _Float16* __restrict__ kbuf,
    const _Float16* __restrict__ vT,
    const float* __restrict__ relh, const float* __restrict__ relw,
    _Float16* __restrict__ obuf) {
  __shared__ __align__(16) _Float16 Qs[128][72];  // A-tile [q][hd]
  __shared__ __align__(16) _Float16 Ks[64][72];   // B-tile [key][hd]
  __shared__ __align__(16) _Float16 Vs[64][72];   // B-tile [hd][key]
  __shared__ __align__(16) _Float16 Ps[128][72];  // A-tile [q][key]

  const int tid   = threadIdx.x;
  const int wave  = tid >> 5;
  const int lane  = tid & 31;
  const int bh    = blockIdx.y;
  const int qbase = blockIdx.x * 128;
  const int nl    = lane & 15;
  const int mb    = (lane >> 4) << 3;
  const int qrow0 = wave * 16;

  // Load Q tile once (128 x 64, row-major, hd contiguous): 1024 chunks, 4 per thread.
#pragma unroll
  for (int c = 0; c < 4; ++c) {
    const int idx = tid * 4 + c;
    const int row = idx >> 3;
    const int cb  = idx & 7;
    const int pb  = (cb & 4) | permblk(cb & 3);
    *(v8h*)(&Qs[row][pb * 8]) =
        *(const v8h*)(qbuf + ((size_t)bh * SEQ + qbase + row) * HDIM + cb * 8);
  }

  float m_prev[8], l_run[8];
  v8f acc_o[4];
#pragma unroll
  for (int r = 0; r < 8; ++r) { m_prev[r] = -1e30f; l_run[r] = 0.f; }
#pragma unroll
  for (int j = 0; j < 4; ++j)
    acc_o[j] = (v8f){0.f, 0.f, 0.f, 0.f, 0.f, 0.f, 0.f, 0.f};

  const float* rh_row = relh + ((size_t)bh * SEQ) * 32;
  const float* rw_row = relw + ((size_t)bh * SEQ) * 32;

  for (int kt = 0; kt < SEQ; kt += 64) {
    // Stage K [key][hd] and V^T [hd][key]: 512 chunks each, 2 per thread.
#pragma unroll
    for (int c = 0; c < 2; ++c) {
      const int idx = tid * 2 + c;   // 0..511
      const int row = idx >> 3;      // 0..63
      const int cb  = idx & 7;
      *(v8h*)(&Ks[row][cb * 8]) =
          *(const v8h*)(kbuf + ((size_t)bh * SEQ + kt + row) * HDIM + cb * 8);
      *(v8h*)(&Vs[row][cb * 8]) =
          *(const v8h*)(vT + ((size_t)bh * HDIM + row) * SEQ + kt + cb * 8);
    }
    __syncthreads();

    // Hoisted rel bias: keys kt..kt+63 span kh in {kt>>5, kt>>5+1},
    // kw = (ct&1)*16 + nl.
    const int kh0 = kt >> 5;
    float rh[2][8], rw[2][8];
#pragma unroll
    for (int r = 0; r < 8; ++r) {
      const size_t qs = (size_t)(qbase + qrow0 + mb + r) * 32;
      rh[0][r] = rh_row[qs + kh0];
      rh[1][r] = rh_row[qs + kh0 + 1];
      rw[0][r] = rw_row[qs + nl];
      rw[1][r] = rw_row[qs + 16 + nl];
    }

    // S = Q K^T : wave = 16 q-rows x 64 keys = 4 C tiles, 2 WMMAs each.
    const v16h qa0 = load_frag(&Qs[qrow0][0], 72);   // hd 0..31
    const v16h qa1 = load_frag(&Qs[qrow0][32], 72);  // hd 32..63
    float sf[4][8];
#pragma unroll
    for (int ct = 0; ct < 4; ++ct) {
      v8f s = (v8f){0.f, 0.f, 0.f, 0.f, 0.f, 0.f, 0.f, 0.f};
      s = wmma_f16(qa0, load_frag(&Ks[ct * 16][0], 72), s);
      s = wmma_f16(qa1, load_frag(&Ks[ct * 16][32], 72), s);
#pragma unroll
      for (int r = 0; r < 8; ++r)
        sf[ct][r] = s[r] * SCALE + rh[ct >> 1][r] + rw[ct & 1][r];
    }

    // Online softmax (row = one VGPR lane-group; reduce with shfl width 16).
    float alpha[8], mnew[8], rsum[8];
#pragma unroll
    for (int r = 0; r < 8; ++r) {
      float mx = fmaxf(fmaxf(sf[0][r], sf[1][r]), fmaxf(sf[2][r], sf[3][r]));
#pragma unroll
      for (int off = 1; off < 16; off <<= 1)
        mx = fmaxf(mx, __shfl_xor(mx, off, 16));
      mnew[r]  = fmaxf(m_prev[r], mx);
      alpha[r] = __expf(m_prev[r] - mnew[r]);
      rsum[r]  = 0.f;
    }
#pragma unroll
    for (int ct = 0; ct < 4; ++ct) {
      const int cc = ct * 16 + nl;   // key column 0..63
      const int pc = cc;
#pragma unroll
      for (int r = 0; r < 8; ++r) {
        const float p = __expf(sf[ct][r] - mnew[r]);
        rsum[r] += p;
        Ps[qrow0 + mb + r][pc] = (_Float16)p;   // A-tile [q][key] position
      }
    }
#pragma unroll
    for (int r = 0; r < 8; ++r) {
      float rs = rsum[r];
#pragma unroll
      for (int off = 1; off < 16; off <<= 1) rs += __shfl_xor(rs, off, 16);
      l_run[r]  = l_run[r] * alpha[r] + rs;
      m_prev[r] = mnew[r];
    }
#pragma unroll
    for (int j = 0; j < 4; ++j)
#pragma unroll
      for (int r = 0; r < 8; ++r) acc_o[j][r] = acc_o[j][r] * alpha[r];

    // O += P V (same-wave LDS round trip; DScnt orders store->load).
    const v16h pa0 = load_frag(&Ps[qrow0][0], 72);   // keys 0..31
    const v16h pa1 = load_frag(&Ps[qrow0][32], 72);  // keys 32..63
#pragma unroll
    for (int j = 0; j < 4; ++j) {
      acc_o[j] = wmma_f16(pa0, load_frag(&Vs[j * 16][0], 72), acc_o[j]);
      acc_o[j] = wmma_f16(pa1, load_frag(&Vs[j * 16][32], 72), acc_o[j]);
    }
    __syncthreads();   // before next tile overwrites Ks/Vs
  }

  // Normalize; store (b, s, head*64+hd) f16 for the proj GEMM.
  const int b    = bh / NHEAD;
  const int head = bh % NHEAD;
#pragma unroll
  for (int j = 0; j < 4; ++j)
#pragma unroll
    for (int r = 0; r < 8; ++r) {
      const int qs = qbase + qrow0 + mb + r;
      const float val = acc_o[j][r] / l_run[r];
      obuf[((size_t)(b * SEQ + qs)) * DIMC + head * HDIM + j * 16 + nl] =
          (_Float16)val;
    }
}

// ---------------------------------------------------------------------------
// Kernel 4: out = obuf @ proj_w^T + proj_b (fp32 out). M=8192,N=768,K=768.
// ---------------------------------------------------------------------------
__global__ __launch_bounds__(256) void proj_gemm_kernel(
    const _Float16* __restrict__ a_in, const float* __restrict__ proj_w,
    const float* __restrict__ proj_b, float* __restrict__ out) {
  __shared__ __align__(16) _Float16 As[128][40];
  __shared__ __align__(16) _Float16 Bs[128][40];

  const int tid  = threadIdx.x;
  const int wave = tid >> 5;
  const int lane = tid & 31;
  const int bm   = blockIdx.y * 128;
  const int bn   = blockIdx.x * 128;
  const int wm   = (wave & 3) * 32;
  const int wn   = (wave >> 2) * 64;

  v8f acc[2][4];
#pragma unroll
  for (int i = 0; i < 2; ++i)
#pragma unroll
    for (int j = 0; j < 4; ++j)
      acc[i][j] = (v8f){0.f, 0.f, 0.f, 0.f, 0.f, 0.f, 0.f, 0.f};

  for (int kt = 0; kt < DIMC; kt += 32) {
#pragma unroll
    for (int c = 0; c < 2; ++c) {
      const int idx = tid * 2 + c;
      const int row = idx >> 2;
      const int kb  = idx & 3;
      const _Float16* ap = a_in + (size_t)(bm + row) * DIMC + kt + kb * 8;
      const float*    wp = proj_w + (size_t)(bn + row) * DIMC + kt + kb * 8;
      if (kt + 32 < DIMC) {
        __builtin_prefetch(ap + 32, 0, 0);
        __builtin_prefetch(wp + 32, 0, 0);
      }
      *(v8h*)(&As[row][permblk(kb) * 8]) = *(const v8h*)ap;  // already f16
      *(v8h*)(&Bs[row][kb * 8]) =
          cvt8(*(const vf4*)wp, *(const vf4*)(wp + 4));
    }
    __syncthreads();

    const v16h a0 = load_frag(&As[wm][0], 40);
    const v16h a1 = load_frag(&As[wm + 16][0], 40);
#pragma unroll
    for (int j = 0; j < 4; ++j) {
      const v16h b = load_frag(&Bs[wn + j * 16][0], 40);
      acc[0][j] = wmma_f16(a0, b, acc[0][j]);
      acc[1][j] = wmma_f16(a1, b, acc[1][j]);
    }
    __syncthreads();
  }

  const int nl = lane & 15;
  const int mb = (lane >> 4) << 3;
#pragma unroll
  for (int i = 0; i < 2; ++i)
#pragma unroll
    for (int j = 0; j < 4; ++j)
#pragma unroll
      for (int r = 0; r < 8; ++r) {
        const int gm = bm + wm + i * 16 + mb + r;
        const int gn = bn + wn + j * 16 + nl;
        out[(size_t)gm * DIMC + gn] = acc[i][j][r] + proj_b[gn];
      }
}

// ---------------------------------------------------------------------------
// Launch. Workspace slabs (12 MiB each):
//   [0] q f16 (96,1024,64)  [1] k f16 (96,1024,64)  [2] v^T f16 (96,64,1024)
//   [3] obuf f16 (8,1024,768)  [4] rel_h f32 (96,1024,32)  [5] rel_w f32
// ---------------------------------------------------------------------------
extern "C" void kernel_launch(void* const* d_in, const int* in_sizes, int n_in,
                              void* d_out, int out_size, void* d_ws,
                              size_t ws_size, hipStream_t stream) {
  (void)in_sizes; (void)n_in; (void)out_size; (void)ws_size;
  const float* x         = (const float*)d_in[0];
  const float* qkv_w     = (const float*)d_in[1];
  const float* qkv_b     = (const float*)d_in[2];
  const float* proj_w    = (const float*)d_in[3];
  const float* proj_b    = (const float*)d_in[4];
  const float* rel_pos_h = (const float*)d_in[5];
  const float* rel_pos_w = (const float*)d_in[6];

  const size_t SLAB = (size_t)12582912;  // 12 MiB
  char* ws = (char*)d_ws;
  _Float16* qbuf = (_Float16*)(ws + 0 * SLAB);
  _Float16* kbuf = (_Float16*)(ws + 1 * SLAB);
  _Float16* vT   = (_Float16*)(ws + 2 * SLAB);
  _Float16* obuf = (_Float16*)(ws + 3 * SLAB);
  float*    relh = (float*)(ws + 4 * SLAB);
  float*    relw = (float*)(ws + 5 * SLAB);

  qkv_gemm_kernel<<<dim3(18, 64), 256, 0, stream>>>(x, qkv_w, qkv_b, qbuf,
                                                    kbuf, vT);
  rel_bias_kernel<<<dim3(BH * SEQ / 4), 256, 0, stream>>>(qbuf, rel_pos_h,
                                                          rel_pos_w, relh,
                                                          relw);
  flash_attn_kernel<<<dim3(8, BH), 256, 0, stream>>>(qbuf, kbuf, vT, relh,
                                                     relw, obuf);
  proj_gemm_kernel<<<dim3(6, 64), 256, 0, stream>>>(obuf, proj_w, proj_b,
                                                    (float*)d_out);
}
